// EncoderBlock_37795712205377
// MI455X (gfx1250) — matrix operand order repaired and checked by reading, not compiled, not run on hardware
//
#include <hip/hip_runtime.h>

// ---------------------------------------------------------------------------
// Transformer encoder block for MI455X (gfx1250, wave32, WMMA).
// All matmuls use v_wmma_f32_16x16x32_bf16 (bf16 operands, f32 accumulate).
// GEMM tiles staged via GLOBAL_LOAD_ASYNC_TO_LDS_B128 (ASYNCcnt-tracked).
// ---------------------------------------------------------------------------

typedef __attribute__((ext_vector_type(16))) __bf16 v16bf;
typedef __attribute__((ext_vector_type(8)))  float  v8f;

// Exact parameter types for the async-to-LDS builtin (from the compiler
// diagnostic: "int __attribute__((__vector_size__(4 * sizeof(int)))) *").
typedef int v4i_vs __attribute__((vector_size(16)));
typedef __attribute__((address_space(1))) v4i_vs g_v4i;   // global
typedef __attribute__((address_space(3))) v4i_vs l_v4i;   // LDS

#ifndef __has_builtin
#define __has_builtin(x) 0
#endif
#if __has_builtin(__builtin_amdgcn_global_load_async_to_lds_b128) && \
    __has_builtin(__builtin_amdgcn_s_wait_asynccnt)
#define HAS_ASYNC_LDS 1
#else
#define HAS_ASYNC_LDS 0
#endif

constexpr int Bc   = 4;
constexpr int Sc   = 1024;
constexpr int Dc   = 1024;
constexpr int Hc   = 16;
constexpr int DKc  = 64;
constexpr int DFFc = 4096;

__device__ __forceinline__ v8f wmma_bf16(v16bf a, v16bf b, v8f c) {
  return __builtin_amdgcn_wmma_f32_16x16x32_bf16(
      /*neg_a=*/false, a, /*neg_b=*/false, b,
      /*c_mod=*/(short)0, c, /*reuse_a=*/false, /*reuse_b=*/false);
}

// 16B global -> LDS copy; async (ASYNCcnt) when available.
__device__ __forceinline__ void cp16_g2l(const __bf16* g, __bf16* l) {
#if HAS_ASYNC_LDS
  __builtin_amdgcn_global_load_async_to_lds_b128((g_v4i*)g, (l_v4i*)l, 0, 0);
#else
  *(uint4*)l = *(const uint4*)g;
#endif
}
__device__ __forceinline__ void async_wait_all() {
#if HAS_ASYNC_LDS
  __builtin_amdgcn_s_wait_asynccnt(0);
#endif
}

// A-fragment (16x32 bf16, MxK). Per ISA 7.12.2: lane(0..15)=row M, half-wave
// selects K-subrange: K = {half*8+0..7, 16+half*8+0..7} -> two 16B LDS loads.
__device__ __forceinline__ v16bf frag_a(const __bf16* t, int ld, int m0, int k0) {
  const int lane = threadIdx.x & 31;
  const int hf = lane >> 4, m = lane & 15;
  const __bf16* p = t + (m0 + m) * ld + k0 + hf * 8;
  union { uint4 u[2]; v16bf v; } u;
  u.u[0] = *(const uint4*)(p);
  u.u[1] = *(const uint4*)(p + 16);
  return u.v;
}

// B-fragment (32x16 bf16, KxN) from an N-major tile tT[n*ld + k]:
// lane(0..15)=col N, K = {half*8+0..7, 16+half*8+0..7} -> two 16B LDS loads.
__device__ __forceinline__ v16bf frag_bT(const __bf16* tT, int ld, int n0, int k0) {
  const int lane = threadIdx.x & 31;
  const int hf = lane >> 4, n = lane & 15;
  const __bf16* p = tT + (n0 + n) * ld + k0 + hf * 8;
  union { uint4 u[2]; v16bf v; } u;
  u.u[0] = *(const uint4*)(p);
  u.u[1] = *(const uint4*)(p + 16);
  return u.v;
}

// ---------------------------------------------------------------------------
// Conversions / weight repacking (run once per launch; coalesced both ways).
// ---------------------------------------------------------------------------
__global__ void cvt_f32_bf16(const float* __restrict__ in, __bf16* __restrict__ out, int n) {
  int i = blockIdx.x * blockDim.x + threadIdx.x;
  if (i < n) out[i] = (__bf16)in[i];
}

// [K,N] f32 row-major -> [N,K] bf16 (tiled LDS transpose, 32x32 tiles).
__global__ __launch_bounds__(256) void transpose_cvt_bf16(
    const float* __restrict__ in, __bf16* __restrict__ out, int Kd, int Nd) {
  __shared__ float tile[32][33];
  const int kb = blockIdx.y * 32, nb = blockIdx.x * 32;
  const int tx = threadIdx.x & 31, ty = threadIdx.x >> 5;
#pragma unroll
  for (int r = ty; r < 32; r += 8)
    tile[r][tx] = in[(size_t)(kb + r) * Nd + nb + tx];
  __syncthreads();
#pragma unroll
  for (int r = ty; r < 32; r += 8)
    out[(size_t)(nb + r) * Kd + kb + tx] = (__bf16)tile[tx][r];
}

// [H, D, DK] f32 -> [(h*DK+dk), D] bf16 (per-head transpose) so the QKV
// projection is one GEMM against an N-major weight.
__global__ __launch_bounds__(256) void repack_headw_T(
    const float* __restrict__ in, __bf16* __restrict__ out) {
  __shared__ float tile[32][33];
  const int h = blockIdx.z;
  const int db = blockIdx.y * 32, kb = blockIdx.x * 32;  // d, dk
  const float* src = in + (size_t)h * Dc * DKc;
  __bf16* dst = out + (size_t)h * DKc * Dc;
  const int tx = threadIdx.x & 31, ty = threadIdx.x >> 5;
#pragma unroll
  for (int r = ty; r < 32; r += 8)
    tile[r][tx] = src[(size_t)(db + r) * DKc + kb + tx];
  __syncthreads();
#pragma unroll
  for (int r = ty; r < 32; r += 8)
    dst[(size_t)(kb + r) * Dc + db + tx] = (__bf16)tile[tx][r];
}

// ---------------------------------------------------------------------------
// Tiled WMMA GEMM: C[M,N] = act(alpha * A[M,K] x BT[N,K]^T + bias[N])
// BM=BN=128, BK=32; 8 waves, wave grid 2(M) x 4(N) -> 64x32 per wave.
// Double-buffered LDS; async global->LDS tile staging.
// ---------------------------------------------------------------------------
#define BM 128
#define BN 128
#define BK 32

__global__ __launch_bounds__(256) void gemm_bf16_wmma(
    const __bf16* __restrict__ Ap, const __bf16* __restrict__ BT,
    const float* __restrict__ bias, float alpha, int relu,
    float* __restrict__ Cf, __bf16* __restrict__ Cb,
    int M, int N, int K) {
  __shared__ __align__(16) __bf16 As[2][BM * BK];   // [m][k]
  __shared__ __align__(16) __bf16 Bs[2][BN * BK];   // [n][k]

  const int tid = threadIdx.x;
  const int lane = tid & 31;
  const int wv = tid >> 5;
  const int wm = wv & 1;     // 2 waves along M
  const int wn = wv >> 1;    // 4 waves along N
  const int m0 = blockIdx.y * BM;
  const int n0 = blockIdx.x * BN;

  // Per-thread tile-fill coordinates: 512 x 16B per tile, 2 per thread.
  const int r0 = (tid * 2) >> 2, s0 = ((tid * 2) & 3) * 8;
  const int r1 = (tid * 2 + 1) >> 2, s1 = ((tid * 2 + 1) & 3) * 8;

  const v8f zero8 = {0.f, 0.f, 0.f, 0.f, 0.f, 0.f, 0.f, 0.f};
  v8f acc[4][2];
#pragma unroll
  for (int i = 0; i < 4; ++i)
#pragma unroll
    for (int j = 0; j < 2; ++j) acc[i][j] = zero8;

  const int kTiles = K / BK;

  // Prologue: stage tile 0 into buffer 0.
  cp16_g2l(Ap + (size_t)(m0 + r0) * K + s0, &As[0][r0 * BK + s0]);
  cp16_g2l(Ap + (size_t)(m0 + r1) * K + s1, &As[0][r1 * BK + s1]);
  cp16_g2l(BT + (size_t)(n0 + r0) * K + s0, &Bs[0][r0 * BK + s0]);
  cp16_g2l(BT + (size_t)(n0 + r1) * K + s1, &Bs[0][r1 * BK + s1]);

  for (int kt = 0; kt < kTiles; ++kt) {
    const int p = kt & 1;
    async_wait_all();
    __syncthreads();

#if !HAS_ASYNC_LDS
    uint4 ra0, ra1, rb0, rb1;
#endif
    if (kt + 1 < kTiles) {
      const int ko = (kt + 1) * BK;
#if HAS_ASYNC_LDS
      cp16_g2l(Ap + (size_t)(m0 + r0) * K + ko + s0, &As[1 - p][r0 * BK + s0]);
      cp16_g2l(Ap + (size_t)(m0 + r1) * K + ko + s1, &As[1 - p][r1 * BK + s1]);
      cp16_g2l(BT + (size_t)(n0 + r0) * K + ko + s0, &Bs[1 - p][r0 * BK + s0]);
      cp16_g2l(BT + (size_t)(n0 + r1) * K + ko + s1, &Bs[1 - p][r1 * BK + s1]);
#else
      ra0 = *(const uint4*)(Ap + (size_t)(m0 + r0) * K + ko + s0);
      ra1 = *(const uint4*)(Ap + (size_t)(m0 + r1) * K + ko + s1);
      rb0 = *(const uint4*)(BT + (size_t)(n0 + r0) * K + ko + s0);
      rb1 = *(const uint4*)(BT + (size_t)(n0 + r1) * K + ko + s1);
#endif
      __builtin_prefetch(Ap + (size_t)(m0 + r0) * K + (kt + 2) * BK + s0, 0, 1);
    }

    v16bf bf[2];
#pragma unroll
    for (int j = 0; j < 2; ++j) bf[j] = frag_bT(Bs[p], BK, wn * 32 + j * 16, 0);
#pragma unroll
    for (int i = 0; i < 4; ++i) {
      v16bf af = frag_a(As[p], BK, wm * 64 + i * 16, 0);
#pragma unroll
      for (int j = 0; j < 2; ++j) acc[i][j] = wmma_bf16(af, bf[j], acc[i][j]);
    }

#if !HAS_ASYNC_LDS
    if (kt + 1 < kTiles) {
      *(uint4*)&As[1 - p][r0 * BK + s0] = ra0;
      *(uint4*)&As[1 - p][r1 * BK + s1] = ra1;
      *(uint4*)&Bs[1 - p][r0 * BK + s0] = rb0;
      *(uint4*)&Bs[1 - p][r1 * BK + s1] = rb1;
    }
#endif
  }

  // Epilogue. C-layout: VGPR r, lanes 0-15 -> row r, lanes 16-31 -> row r+8.
  const int hf = lane >> 4, ln = lane & 15;
#pragma unroll
  for (int i = 0; i < 4; ++i) {
#pragma unroll
    for (int j = 0; j < 2; ++j) {
      const int col = n0 + wn * 32 + j * 16 + ln;
      const float bv = bias ? bias[col] : 0.f;
#pragma unroll
      for (int r = 0; r < 8; ++r) {
        const int row = m0 + wm * 64 + i * 16 + r + hf * 8;
        float v = acc[i][j][r] * alpha + bv;
        if (relu) v = fmaxf(v, 0.f);
        if (Cf) Cf[(size_t)row * N + col] = v;
        if (Cb) Cb[(size_t)row * N + col] = (__bf16)v;
      }
    }
  }
}

// ---------------------------------------------------------------------------
// Fused flash attention (eval mode). Q/K/V bf16 in [B,S,H,DK] layout.
// Grid: (S/128, B*H); wave w owns query rows 16w..16w+15 so online-softmax
// stats reduce inside one 16-lane group (wave32). Scale folded into Q.
// ---------------------------------------------------------------------------
#define AQ 128
#define AKB 64

__global__ __launch_bounds__(256) void attn_wmma(
    const __bf16* __restrict__ Q, const __bf16* __restrict__ Kb,
    const __bf16* __restrict__ Vb, __bf16* __restrict__ O) {
  __shared__ __align__(16) __bf16 Qs[AQ * DKc];     // [qrow][dk]
  __shared__ __align__(16) __bf16 Ks[AKB * DKc];    // [key][dk]  (= B^T for QK^T)
  __shared__ __align__(16) __bf16 VsT[DKc * AKB];   // [dk][key]  (= B^T for PV)
  __shared__ __align__(16) __bf16 Ps[AQ * AKB];     // [qrow][key]

  const int tid = threadIdx.x, lane = tid & 31, wv = tid >> 5;
  const int bh = blockIdx.y, b = bh >> 4, h = bh & 15;
  const int q0 = blockIdx.x * AQ;
  const size_t rowStride = (size_t)Hc * DKc;  // 1024

  const __bf16* Qg = Q + ((size_t)(b * Sc + q0) * Hc + h) * DKc;
#pragma unroll
  for (int u = 0; u < 4; ++u) {  // 128x64 = 1024 x 16B, 4 per thread
    int i = tid * 4 + u;
    int r = i >> 3, seg = i & 7;
    cp16_g2l(Qg + (size_t)r * rowStride + seg * 8, &Qs[r * DKc + seg * 8]);
  }

  const v8f zero8 = {0.f, 0.f, 0.f, 0.f, 0.f, 0.f, 0.f, 0.f};
  float mi[8], li[8];
  v8f oacc[4];
#pragma unroll
  for (int r = 0; r < 8; ++r) { mi[r] = -INFINITY; li[r] = 0.f; }
#pragma unroll
  for (int j = 0; j < 4; ++j) oacc[j] = zero8;

  for (int kt = 0; kt < Sc / AKB; ++kt) {
    const __bf16* Kg = Kb + ((size_t)(b * Sc + kt * AKB) * Hc + h) * DKc;
    const __bf16* Vg = Vb + ((size_t)(b * Sc + kt * AKB) * Hc + h) * DKc;
    __syncthreads();   // prior iteration's reads of Ks/VsT done
#pragma unroll
    for (int u = 0; u < 2; ++u) {  // K tile 64x64 = 512 x 16B (async path)
      int i = tid * 2 + u;
      int r = i >> 3, seg = i & 7;
      cp16_g2l(Kg + (size_t)r * rowStride + seg * 8, &Ks[r * DKc + seg * 8]);
    }
#pragma unroll
    for (int u = 0; u < 2; ++u) {  // V tile, stored transposed [dk][key]
      int i = tid * 2 + u;
      int r = i >> 3, seg = i & 7;
      uint4 d = *(const uint4*)(Vg + (size_t)r * rowStride + seg * 8);
      const __bf16* dv = (const __bf16*)&d;
#pragma unroll
      for (int e = 0; e < 8; ++e) VsT[(seg * 8 + e) * AKB + r] = dv[e];
    }
    async_wait_all();
    __syncthreads();

    // S = Q K^T for this wave's 16 rows x 64 keys.
    v8f sa[4];
#pragma unroll
    for (int j = 0; j < 4; ++j) sa[j] = zero8;
#pragma unroll
    for (int ks = 0; ks < 2; ++ks) {
      v16bf aq = frag_a(Qs, DKc, wv * 16, ks * 32);
#pragma unroll
      for (int j = 0; j < 4; ++j) {
        v16bf bk = frag_bT(Ks, DKc, j * 16, ks * 32);
        sa[j] = wmma_bf16(aq, bk, sa[j]);
      }
    }

    // Online softmax: row r+8*half lives in one 16-lane group -> shfl width 16.
#pragma unroll
    for (int r = 0; r < 8; ++r) {
      float mrow = -INFINITY;
#pragma unroll
      for (int j = 0; j < 4; ++j) mrow = fmaxf(mrow, sa[j][r]);
#pragma unroll
      for (int off = 1; off < 16; off <<= 1) mrow = fmaxf(mrow, __shfl_xor(mrow, off, 16));
      const float mnew = fmaxf(mi[r], mrow);
      const float corr = __expf(mi[r] - mnew);
      float psum = 0.f;
#pragma unroll
      for (int j = 0; j < 4; ++j) {
        float p = __expf(sa[j][r] - mnew);
        sa[j][r] = p;
        psum += p;
      }
#pragma unroll
      for (int off = 1; off < 16; off <<= 1) psum += __shfl_xor(psum, off, 16);
      li[r] = li[r] * corr + psum;
      mi[r] = mnew;
#pragma unroll
      for (int j = 0; j < 4; ++j) oacc[j][r] *= corr;
    }

    // Stage P (C-layout -> LDS -> A-layout). Wave-private rows; same-wave DS
    // ops are in-order so no barrier needed between store and reload.
    {
      const int hf = lane >> 4, ln = lane & 15;
#pragma unroll
      for (int j = 0; j < 4; ++j)
#pragma unroll
        for (int r = 0; r < 8; ++r)
          Ps[(wv * 16 + r + hf * 8) * AKB + j * 16 + ln] = (__bf16)sa[j][r];
    }

    // O += P V
#pragma unroll
    for (int ks = 0; ks < 2; ++ks) {
      v16bf ap = frag_a(Ps, AKB, wv * 16, ks * 32);
#pragma unroll
      for (int j = 0; j < 4; ++j) {
        v16bf bvf = frag_bT(VsT, AKB, j * 16, ks * 32);
        oacc[j] = wmma_bf16(ap, bvf, oacc[j]);
      }
    }
  }

  // Write O (head-concat layout [B,S,H*DK]) as bf16 for the Wo GEMM.
  const int hf = lane >> 4, ln = lane & 15;
#pragma unroll
  for (int j = 0; j < 4; ++j)
#pragma unroll
    for (int r = 0; r < 8; ++r) {
      const int s = q0 + wv * 16 + r + hf * 8;
      const float v = oacc[j][r] / li[r];
      O[((size_t)(b * Sc + s)) * (Hc * DKc) + h * DKc + j * 16 + ln] = (__bf16)v;
    }
}

// ---------------------------------------------------------------------------
// y = res + LayerNorm(a) * g + be    (population variance, eps=1e-5)
// ---------------------------------------------------------------------------
__global__ __launch_bounds__(256) void ln_residual(
    const float* __restrict__ res, const float* __restrict__ a,
    const float* __restrict__ g, const float* __restrict__ be,
    float* __restrict__ outf, __bf16* __restrict__ outb) {
  const int row = blockIdx.x;
  const int tid = threadIdx.x;
  const float* ar = a + (size_t)row * Dc;

  float s = 0.f, ss = 0.f;
  for (int i = tid; i < Dc; i += 256) {
    float v = ar[i];
    s += v;
    ss += v * v;
  }
#pragma unroll
  for (int off = 1; off < 32; off <<= 1) {
    s += __shfl_xor(s, off, 32);
    ss += __shfl_xor(ss, off, 32);
  }
  __shared__ float rs[8], rss[8];
  if ((tid & 31) == 0) { rs[tid >> 5] = s; rss[tid >> 5] = ss; }
  __syncthreads();
  float S = 0.f, SS = 0.f;
#pragma unroll
  for (int wv = 0; wv < 8; ++wv) { S += rs[wv]; SS += rss[wv]; }
  const float mean = S * (1.f / Dc);
  const float var = SS * (1.f / Dc) - mean * mean;
  const float rstd = rsqrtf(var + 1e-5f);

  for (int i = tid; i < Dc; i += 256) {
    float v = res[(size_t)row * Dc + i] + (ar[i] - mean) * rstd * g[i] + be[i];
    outf[(size_t)row * Dc + i] = v;
    if (outb) outb[(size_t)row * Dc + i] = (__bf16)v;
  }
}

// ---------------------------------------------------------------------------
// Host orchestration
// ---------------------------------------------------------------------------
extern "C" void kernel_launch(void* const* d_in, const int* in_sizes, int n_in,
                              void* d_out, int out_size, void* d_ws, size_t ws_size,
                              hipStream_t stream) {
  (void)in_sizes; (void)n_in; (void)out_size; (void)ws_size;

  const float* x   = (const float*)d_in[0];
  const float* Wq  = (const float*)d_in[1];
  const float* bq  = (const float*)d_in[2];
  const float* Wk  = (const float*)d_in[3];
  const float* bk  = (const float*)d_in[4];
  const float* Wv  = (const float*)d_in[5];
  const float* bv  = (const float*)d_in[6];
  const float* Wo  = (const float*)d_in[7];
  const float* bo  = (const float*)d_in[8];
  const float* W1  = (const float*)d_in[9];
  const float* b1  = (const float*)d_in[10];
  const float* W2  = (const float*)d_in[11];
  const float* b2  = (const float*)d_in[12];
  const float* g1  = (const float*)d_in[13];
  const float* be1 = (const float*)d_in[14];
  const float* g2  = (const float*)d_in[15];
  const float* be2 = (const float*)d_in[16];

  const size_t BSD = (size_t)Bc * Sc * Dc;        // 4,194,304
  const size_t DHK = (size_t)Dc * Hc * DKc;       // 1,048,576
  const size_t DW1 = (size_t)Dc * DFFc;           // 4,194,304
  const size_t BSF = (size_t)Bc * Sc * DFFc;      // 16,777,216

  char* w = (char*)d_ws;
  size_t off = 0;
  auto carve = [&](size_t bytes) -> void* {
    void* p = w + off;
    off += (bytes + 255) & ~(size_t)255;
    return p;
  };

  __bf16* xb   = (__bf16*)carve(BSD * 2);
  __bf16* wqbT = (__bf16*)carve(DHK * 2);   // [(h,dk), D]
  __bf16* wkbT = (__bf16*)carve(DHK * 2);
  __bf16* wvbT = (__bf16*)carve(DHK * 2);
  __bf16* wobT = (__bf16*)carve(DHK * 2);   // [D, H*DK]
  __bf16* w1bT = (__bf16*)carve(DW1 * 2);   // [DFF, D]
  __bf16* w2bT = (__bf16*)carve(DW1 * 2);   // [D, DFF]
  __bf16* qb   = (__bf16*)carve(BSD * 2);
  __bf16* kbb  = (__bf16*)carve(BSD * 2);
  __bf16* vbb  = (__bf16*)carve(BSD * 2);
  __bf16* ob   = (__bf16*)carve(BSD * 2);
  float*  attf = (float*)carve(BSD * 4);
  float*  x1f  = (float*)carve(BSD * 4);
  __bf16* x1b  = (__bf16*)carve(BSD * 2);
  __bf16* h1b  = (__bf16*)carve(BSF * 2);
  float*  posf = (float*)carve(BSD * 4);

  const int T = 256;
  // --- precision conversion / transposed weight repacking ---
  cvt_f32_bf16<<<(int)(BSD / T), T, 0, stream>>>(x, xb, (int)BSD);
  {
    dim3 gqkv(DKc / 32, Dc / 32, Hc);
    repack_headw_T<<<gqkv, T, 0, stream>>>(Wq, wqbT);
    repack_headw_T<<<gqkv, T, 0, stream>>>(Wk, wkbT);
    repack_headw_T<<<gqkv, T, 0, stream>>>(Wv, wvbT);
  }
  transpose_cvt_bf16<<<dim3(Dc / 32, Dc / 32), T, 0, stream>>>(Wo, wobT, Dc, Dc);
  transpose_cvt_bf16<<<dim3(DFFc / 32, Dc / 32), T, 0, stream>>>(W1, w1bT, Dc, DFFc);
  transpose_cvt_bf16<<<dim3(Dc / 32, DFFc / 32), T, 0, stream>>>(W2, w2bT, DFFc, Dc);

  const int M = Bc * Sc;  // 4096
  // --- QKV projections (scale folded into Q) ---
  gemm_bf16_wmma<<<dim3(Dc / BN, M / BM), T, 0, stream>>>(
      xb, wqbT, bq, 0.125f, 0, nullptr, qb, M, Dc, Dc);
  gemm_bf16_wmma<<<dim3(Dc / BN, M / BM), T, 0, stream>>>(
      xb, wkbT, bk, 1.0f, 0, nullptr, kbb, M, Dc, Dc);
  gemm_bf16_wmma<<<dim3(Dc / BN, M / BM), T, 0, stream>>>(
      xb, wvbT, bv, 1.0f, 0, nullptr, vbb, M, Dc, Dc);

  // --- fused flash attention ---
  attn_wmma<<<dim3(Sc / AQ, Bc * Hc), T, 0, stream>>>(qb, kbb, vbb, ob);

  // --- output projection ---
  gemm_bf16_wmma<<<dim3(Dc / BN, M / BM), T, 0, stream>>>(
      ob, wobT, bo, 1.0f, 0, attf, nullptr, M, Dc, Dc);

  // --- x1 = x + LN(att) ---
  ln_residual<<<M, T, 0, stream>>>(x, attf, g1, be1, x1f, x1b);

  // --- FFN ---
  gemm_bf16_wmma<<<dim3(DFFc / BN, M / BM), T, 0, stream>>>(
      x1b, w1bT, b1, 1.0f, 1, nullptr, h1b, M, DFFc, Dc);
  gemm_bf16_wmma<<<dim3(Dc / BN, M / BM), T, 0, stream>>>(
      h1b, w2bT, b2, 1.0f, 0, posf, nullptr, M, Dc, DFFc);

  // --- out = x1 + LN(pos) ---
  ln_residual<<<M, T, 0, stream>>>(x1f, posf, g2, be2, (float*)d_out, nullptr);
}
